// GCLSTM_RecurrentGCN_16192026706535
// MI455X (gfx1250) — matrix operand order repaired and checked by reading, not compiled, not run on hardware
//
#include <hip/hip_runtime.h>

typedef __attribute__((ext_vector_type(16))) __bf16 v16bf;
typedef __attribute__((ext_vector_type(8)))  float  v8f;

#define FDIM 256
#define KCAT 768          // 3 * FDIM (Tx0 | Tx1 | Tx2)
#define CDIM 1024         // 4 gates * FDIM
#define KT32 24           // KCAT / 32

// ---------------------------------------------------------------- utilities
__global__ void k_zero(float* p, long n) {
  long i = (long)blockIdx.x * blockDim.x + threadIdx.x;
  if (i < n) p[i] = 0.0f;
}

__global__ void k_init_neg(const float* __restrict__ h, float* __restrict__ dst, long n) {
  long i = (long)blockIdx.x * blockDim.x + threadIdx.x;
  if (i < n) dst[i] = -h[i];
}

// deg[src[e]] += w[e]
__global__ void k_deg(const int* __restrict__ src, const float* __restrict__ w,
                      float* __restrict__ deg, int E) {
  int e = blockIdx.x * blockDim.x + threadIdx.x;
  if (e < E) atomicAdd(&deg[src[e]], w[e]);
}

__global__ void k_dis(const float* __restrict__ deg, float* __restrict__ dis, int N) {
  int i = blockIdx.x * blockDim.x + threadIdx.x;
  if (i < N) {
    float d = deg[i];
    dis[i] = (d > 0.0f) ? rsqrtf(fmaxf(d, 1e-12f)) : 0.0f;
  }
}

__global__ void k_normw(const int* __restrict__ src, const int* __restrict__ dst,
                        const float* __restrict__ w, const float* __restrict__ dis,
                        float* __restrict__ nw, int E) {
  int e = blockIdx.x * blockDim.x + threadIdx.x;
  if (e < E) nw[e] = dis[src[e]] * w[e] * dis[dst[e]];
}

// Zdst[dst[e], :] += coef * nw[e] * Zsrc[src[e], :]    (64 lanes/edge, float4)
__global__ void k_scatter(const int* __restrict__ src, const int* __restrict__ dst,
                          const float* __restrict__ nw, const float* __restrict__ Zs,
                          float* __restrict__ Zd, float coef, int E) {
  int t = threadIdx.x;
  int e = blockIdx.x * 4 + (t >> 6);
  int f4 = t & 63;
  if (e >= E) return;
  int s = src[e], d = dst[e];
  float wv = coef * nw[e];
  const float4* srow = (const float4*)(Zs + (long)s * FDIM);
  float4 v = srow[f4];
  float* drow = Zd + (long)d * FDIM + f4 * 4;
  atomicAdd(drow + 0, wv * v.x);
  atomicAdd(drow + 1, wv * v.y);
  atomicAdd(drow + 2, wv * v.z);
  atomicAdd(drow + 3, wv * v.w);
}

// ---------------------------------------------------------------- WMMA packing
// A tile (16x32 bf16) register layout (ISA 7.12.2): lane = m + 16*b where
// b = (k>>3)&1 ; element e = (k&7) + 8*(k>>4).
__global__ void k_packA(const float* __restrict__ h, const float* __restrict__ t1,
                        const float* __restrict__ t2, __bf16* __restrict__ Ap, int N) {
  long i = (long)blockIdx.x * blockDim.x + threadIdx.x;
  long total = (long)N * KCAT;
  if (i >= total) return;
  int n = (int)(i / KCAT);
  int k = (int)(i % KCAT);
  float v;
  if (k < FDIM)            v = h [(long)n * FDIM + k];
  else if (k < 2 * FDIM)   v = t1[(long)n * FDIM + (k - FDIM)];
  else                     v = t2[(long)n * FDIM + (k - 2 * FDIM)];
  int tr = n >> 4, m = n & 15;
  int kt = k >> 5, kl = k & 31;
  int b  = (kl >> 3) & 1;
  int e  = (kl & 7) + ((kl >> 4) << 3);
  int lane = m + (b << 4);
  long off = (((long)tr * KT32 + kt) * 32 + lane) * 16 + e;
  Ap[off] = (__bf16)v;
}

// B tile (32x16 bf16): lane = n + 16*((k>>3)&1), element = (k&7) + 8*(k>>4).
// B'[kk, cc] = Theta[g=cc/256][j=kk/256][kk%256][cc%256]
__global__ void k_packB(const float* __restrict__ Theta, __bf16* __restrict__ Bp) {
  long i = (long)blockIdx.x * blockDim.x + threadIdx.x;
  if (i >= (long)KCAT * CDIM) return;
  int kk = (int)(i / CDIM);
  int cc = (int)(i % CDIM);
  int j = kk >> 8, kp = kk & 255;
  int g = cc >> 8, f  = cc & 255;
  float v = Theta[(((long)(g * 3 + j) * FDIM) + kp) * FDIM + f];
  int kt = kk >> 5, kl = kk & 31;
  int ct = cc >> 4, lane = (cc & 15) + (((kl >> 3) & 1) << 4), e = (kl & 7) + ((kl >> 4) << 3);
  long off = (((long)kt * (CDIM / 16) + ct) * 32 + lane) * 16 + e;
  Bp[off] = (__bf16)v;
}

// ---------------------------------------------------------------- WMMA GEMM
// Z (N x 1024) = Apack (N x 768, bf16) @ Bpack (768 x 1024, bf16), f32 accum.
// One wave computes a 32x64 tile: 8 accumulators, 24 k-steps, 192 WMMAs.
// Per k-step: 6 v16bf loads feed 8 WMMAs; all load addresses are
// base + constant (k-stride fits the signed 24-bit instruction offset).
__global__ void __launch_bounds__(256)
k_gemm(const __bf16* __restrict__ Ap, const __bf16* __restrict__ Bp,
       float* __restrict__ Z, int N) {
  int lane = threadIdx.x & 31;
  int wave = threadIdx.x >> 5;
  int wg   = blockIdx.x * 8 + wave;
  int RT   = N >> 5;                  // row tiles of 32
  if (wg >= RT * (CDIM / 64)) return;
  int rt = wg % RT;
  int ct = wg / RT;                   // 0..15 (64-col tiles)

  // per-lane base pointers (v16bf = one lane's 16 bf16 fragment)
  const v16bf* a0p = (const v16bf*)Ap + ((long)(2 * rt)     * KT32) * 32 + lane;
  const v16bf* a1p = (const v16bf*)Ap + ((long)(2 * rt + 1) * KT32) * 32 + lane;
  const v16bf* bp  = (const v16bf*)Bp + (long)(4 * ct) * 32 + lane;

  v8f acc[2][4];
#pragma unroll
  for (int i = 0; i < 2; ++i)
#pragma unroll
    for (int j = 0; j < 4; ++j) acc[i][j] = v8f{};

#pragma unroll
  for (int kt = 0; kt < KT32; ++kt) {
    v16bf a0 = a0p[kt * 32];                       // +1 KiB per step
    v16bf a1 = a1p[kt * 32];
    v16bf b0 = bp[(long)kt * 2048];                // +64 KiB per step
    v16bf b1 = bp[(long)kt * 2048 + 32];
    v16bf b2 = bp[(long)kt * 2048 + 64];
    v16bf b3 = bp[(long)kt * 2048 + 96];
    acc[0][0] = __builtin_amdgcn_wmma_f32_16x16x32_bf16(false, a0, false, b0, (short)0, acc[0][0], false, false);
    acc[0][1] = __builtin_amdgcn_wmma_f32_16x16x32_bf16(false, a0, false, b1, (short)0, acc[0][1], false, false);
    acc[0][2] = __builtin_amdgcn_wmma_f32_16x16x32_bf16(false, a0, false, b2, (short)0, acc[0][2], false, false);
    acc[0][3] = __builtin_amdgcn_wmma_f32_16x16x32_bf16(false, a0, false, b3, (short)0, acc[0][3], false, false);
    acc[1][0] = __builtin_amdgcn_wmma_f32_16x16x32_bf16(false, a1, false, b0, (short)0, acc[1][0], false, false);
    acc[1][1] = __builtin_amdgcn_wmma_f32_16x16x32_bf16(false, a1, false, b1, (short)0, acc[1][1], false, false);
    acc[1][2] = __builtin_amdgcn_wmma_f32_16x16x32_bf16(false, a1, false, b2, (short)0, acc[1][2], false, false);
    acc[1][3] = __builtin_amdgcn_wmma_f32_16x16x32_bf16(false, a1, false, b3, (short)0, acc[1][3], false, false);
  }

  // C/D layout: vgpr r, lane l -> (M = r + 8*(l>=16), Ncol = l%16)
  int rowBase = rt * 32;
  int colBase = ct * 64;
  int half = (lane >> 4) << 3;
  int nn   = lane & 15;
#pragma unroll
  for (int i = 0; i < 2; ++i) {
    float* zb = Z + (long)(rowBase + 16 * i + half) * CDIM + colBase + nn;
#pragma unroll
    for (int j = 0; j < 4; ++j) {
#pragma unroll
      for (int r = 0; r < 8; ++r)
        zb[(long)r * CDIM + 16 * j] = acc[i][j][r];
    }
  }
}

// ---------------------------------------------------------------- LSTM epilogue
// One block (256 threads) per node: gates + readout reduction.
__global__ void __launch_bounds__(256)
k_lstm(const float* __restrict__ Z, const float* __restrict__ x,
       const float* __restrict__ c, const float* __restrict__ Wx,
       const float* __restrict__ b, const float* __restrict__ wc,
       const float* __restrict__ chebb, const float* __restrict__ linw,
       const float* __restrict__ linb, float* __restrict__ out,
       float* __restrict__ outH, float* __restrict__ outC, int N) {
  __shared__ float xs[8];
  __shared__ float red[256];
  int n = blockIdx.x;
  int f = threadIdx.x;
  if (f < 8) xs[f] = x[(long)n * 8 + f];
  __syncthreads();

  float z[4];
#pragma unroll
  for (int g = 0; g < 4; ++g) {
    float acc = Z[(long)n * CDIM + g * FDIM + f] + chebb[g * FDIM + f] + b[g * FDIM + f];
#pragma unroll
    for (int t = 0; t < 8; ++t)
      acc = fmaf(xs[t], Wx[(g * 8 + t) * FDIM + f], acc);
    z[g] = acc;
  }

  float cv = c[(long)n * FDIM + f];
  float I  = 1.0f / (1.0f + expf(-(z[0] + wc[0 * FDIM + f] * cv)));
  float Fg = 1.0f / (1.0f + expf(-(z[1] + wc[1 * FDIM + f] * cv)));
  float T  = tanhf(z[2]);
  float C  = Fg * cv + I * T;
  float O  = 1.0f / (1.0f + expf(-(z[3] + wc[2 * FDIM + f] * C)));
  float H0 = O * tanhf(C);

  outH[(long)n * FDIM + f] = H0;
  outC[(long)n * FDIM + f] = C;

  red[f] = fmaxf(H0, 0.0f) * linw[f];
  __syncthreads();
#pragma unroll
  for (int s = 128; s > 0; s >>= 1) {
    if (f < s) red[f] += red[f + s];
    __syncthreads();
  }
  if (f == 0) out[n] = red[0] + linb[0];
}

// ---------------------------------------------------------------- launcher
extern "C" void kernel_launch(void* const* d_in, const int* in_sizes, int n_in,
                              void* d_out, int out_size, void* d_ws, size_t ws_size,
                              hipStream_t stream) {
  const float* x   = (const float*)d_in[0];
  const int*   ei  = (const int*)  d_in[1];
  const float* ew  = (const float*)d_in[2];
  const float* h   = (const float*)d_in[3];
  const float* c   = (const float*)d_in[4];
  const float* Wx  = (const float*)d_in[5];
  const float* b   = (const float*)d_in[6];
  const float* wc  = (const float*)d_in[7];
  const float* Th  = (const float*)d_in[8];
  const float* cb  = (const float*)d_in[9];
  const float* lw  = (const float*)d_in[10];
  const float* lb  = (const float*)d_in[11];

  const int E = in_sizes[2];
  const int N = in_sizes[3] / FDIM;
  const int* src = ei;
  const int* dst = ei + E;

  // workspace carve-up (256B aligned)
  char* ws = (char*)d_ws;
  size_t off = 0;
  auto take = [&](size_t bytes) { void* p = ws + off; off = (off + bytes + 255) & ~(size_t)255; return p; };
  float*  deg  = (float*) take((size_t)N * 4);
  float*  dis  = (float*) take((size_t)N * 4);
  float*  nw   = (float*) take((size_t)E * 4);
  float*  Tx1  = (float*) take((size_t)N * FDIM * 4);
  float*  Tx2  = (float*) take((size_t)N * FDIM * 4);
  __bf16* Ap   = (__bf16*)take((size_t)N * KCAT * 2);
  __bf16* Bp   = (__bf16*)take((size_t)KCAT * CDIM * 2);
  float*  Z    = (float*) take((size_t)N * CDIM * 4);

  float* out  = (float*)d_out;
  float* outH = out + N;
  float* outC = out + N + (long)N * FDIM;

  long nf = (long)N * FDIM;
  k_zero    <<<dim3((N + 255) / 256),             256, 0, stream>>>(deg, N);
  k_zero    <<<dim3((unsigned)((nf + 255) / 256)),256, 0, stream>>>(Tx1, nf);
  k_init_neg<<<dim3((unsigned)((nf + 255) / 256)),256, 0, stream>>>(h, Tx2, nf);

  k_deg  <<<dim3((E + 255) / 256), 256, 0, stream>>>(src, ew, deg, E);
  k_dis  <<<dim3((N + 255) / 256), 256, 0, stream>>>(deg, dis, N);
  k_normw<<<dim3((E + 255) / 256), 256, 0, stream>>>(src, dst, ew, dis, nw, E);

  // Tx1 = -A_norm h ; Tx2 = -2 A_norm Tx1 - h
  k_scatter<<<dim3((E + 3) / 4), 256, 0, stream>>>(src, dst, nw, h,   Tx1, -1.0f, E);
  k_scatter<<<dim3((E + 3) / 4), 256, 0, stream>>>(src, dst, nw, Tx1, Tx2, -2.0f, E);

  long totA = (long)N * KCAT;
  k_packA<<<dim3((unsigned)((totA + 255) / 256)), 256, 0, stream>>>(h, Tx1, Tx2, Ap, N);
  long totB = (long)KCAT * CDIM;
  k_packB<<<dim3((unsigned)((totB + 255) / 256)), 256, 0, stream>>>(Th, Bp);

  int waves = (N / 32) * (CDIM / 64);
  k_gemm<<<dim3((waves + 7) / 8), 256, 0, stream>>>(Ap, Bp, Z, N);

  k_lstm<<<dim3(N), 256, 0, stream>>>(Z, x, c, Wx, b, wc, cb, lw, lb, out, outH, outC, N);
}
